// TransformerBlock_56272661512361
// MI455X (gfx1250) — matrix operand order repaired and checked by reading, not compiled, not run on hardware
//
#include <hip/hip_runtime.h>
#include <hip/hip_bf16.h>

// ---------------- problem constants (match reference) ----------------
constexpr int B_ = 4, S_ = 2048, D_ = 2048, H_ = 16, HD_ = 128, HID_ = 5632;
constexpr int M_ = B_ * S_;   // 8192 activation rows

// ---------------- CDNA5 WMMA / TDM types ----------------
typedef __attribute__((ext_vector_type(16))) __bf16       v16bf;
typedef __attribute__((ext_vector_type(8)))  float        v8f;
typedef __attribute__((ext_vector_type(4)))  unsigned int v4u;
typedef __attribute__((ext_vector_type(4)))  unsigned int u32x4;
typedef __attribute__((ext_vector_type(8)))  int          i32x8;
typedef __attribute__((ext_vector_type(4)))  int          i32x4;

#if __has_builtin(__builtin_amdgcn_tensor_load_to_lds)
#define HAVE_TDM 1
#if __has_include(<hip/amd_detail/amd_gfx1250_TDM.h>)
#define HAVE_TDM6 1   // therock-10.0 headers -> 6-arg builtin
#endif
#endif

union FragBF { v4u u[2]; v16bf v; };

__device__ __forceinline__ __bf16 f2bf(float f) {
  unsigned u = __builtin_bit_cast(unsigned, f);
  unsigned r = (u + 0x7FFFu + ((u >> 16) & 1u)) >> 16;   // RNE
  return __builtin_bit_cast(__bf16, (unsigned short)r);
}
__device__ __forceinline__ float bf2f(__bf16 b) {
  unsigned short s = __builtin_bit_cast(unsigned short, b);
  return __builtin_bit_cast(float, (unsigned)s << 16);
}
__device__ __forceinline__ v8f wmma_bf16(v16bf a, v16bf b, v8f c) {
  return __builtin_amdgcn_wmma_f32_16x16x32_bf16(
      false, a, false, b, (short)0, c, false, false);
}

// 16-bit A fragment (16x32 MxK): lanes 0-15 row M=lane, K chunks {0..7,16..23};
// lanes 16-31 same rows, K chunks {8..15,24..31}.  base is [row][K], `stride` in elems.
__device__ __forceinline__ v16bf frag_a(const __bf16* base, int stride, int row0,
                                        int k0, int lane) {
  const __bf16* p = base + (size_t)(row0 + (lane & 15)) * stride + k0 + ((lane >> 4) << 3);
  FragBF f;
  f.u[0] = *(const v4u*)p;
  f.u[1] = *(const v4u*)(p + 16);
  return f.v;
}
// 16-bit B fragment (32x16 KxN), base stored as [N][K]: lanes 0-15 col N=lane, K chunks {0..7,16..23};
// lanes 16-31 K chunks {8..15,24..31} (same k order as frag_a).
__device__ __forceinline__ v16bf frag_b(const __bf16* base, int stride, int col0,
                                        int k0, int lane) {
  const __bf16* p = base + (size_t)(col0 + (lane & 15)) * stride + k0 + ((lane >> 4) << 3);
  FragBF f;
  f.u[0] = *(const v4u*)p;
  f.u[1] = *(const v4u*)(p + 16);
  return f.v;
}

#if HAVE_TDM
// TDM: DMA one 128x32 bf16 tile (row stride K elems) into LDS at byte offset lds_off,
// padding each 64B row with 16B so the LDS image is the [128][40] layout frag_a/b expect.
// D# per CDNA5 ISA: group0 = {flags, lds_addr, global_addr, type=2},
// group1 = {data_size=2B | pad(16dw interval, 4dw amount), dims/strides}.
__device__ __forceinline__ void tdm_load_tile(const __bf16* gsrc, unsigned lds_off,
                                              int K) {
  const unsigned long long ga = (unsigned long long)(size_t)gsrc;
  u32x4 g0;
  g0[0] = 1u;                                              // count=1, user mode
  g0[1] = lds_off;                                         // lds_addr (bytes)
  g0[2] = (unsigned)ga;                                    // global_addr[31:0]
  g0[3] = (unsigned)((ga >> 32) & 0x01FFFFFFu) | (2u << 30); // addr[56:32] | type=2
  i32x8 g1;
  g1[0] = (int)((1u << 16)      // data_size = 2 bytes
              | (1u << 20)      // pad_enable
              | (3u << 22)      // pad_interval: 16 DWORDs (64B = one 32-elem row)
              | (3u << 25));    // pad_amount: 4 DWORDs (16B = 8 bf16 pad)
  g1[1] = (int)((unsigned)K << 16);        // tensor_dim0[15:0] = K
  g1[2] = 0;                               // tensor_dim0 hi, tensor_dim1 lo
  g1[3] = (int)(16u | (32u << 16));        // tensor_dim1 = 1<<20, tile_dim0 = 32
  g1[4] = 128;                             // tile_dim1 = 128, tile_dim2 = 0
  g1[5] = K;                               // tensor_dim0_stride = K elems
  g1[6] = 0;
  g1[7] = 0;
  i32x4 z4 = {};
#if HAVE_TDM6
  i32x8 z8 = {};
  __builtin_amdgcn_tensor_load_to_lds(g0, g1, z4, z4, z8, 0);
#else
  __builtin_amdgcn_tensor_load_to_lds(g0, g1, z4, z4, 0);
#endif
}
#endif  // HAVE_TDM

// ---------------- RMSNorm (fp32 rows -> bf16 rows) ----------------
__global__ __launch_bounds__(256) void k_rmsnorm_bf16(const float* __restrict__ x,
                                                      const float* __restrict__ w,
                                                      __bf16* __restrict__ out) {
  __shared__ float red[256];
  const int row = blockIdx.x;
  const float* xr = x + (size_t)row * D_;
  float ss = 0.f;
  for (int i = threadIdx.x; i < D_; i += 256) { float v = xr[i]; ss += v * v; }
  red[threadIdx.x] = ss;
  __syncthreads();
  for (int s = 128; s > 0; s >>= 1) {
    if (threadIdx.x < s) red[threadIdx.x] += red[threadIdx.x + s];
    __syncthreads();
  }
  const float inv = rsqrtf(red[0] * (1.0f / (float)D_) + 1e-6f);
  __bf16* orow = out + (size_t)row * D_;
  for (int i = threadIdx.x; i < D_; i += 256) orow[i] = f2bf(xr[i] * inv * w[i]);
}

// ---------------- fp32 -> bf16 convert (weights) ----------------
__global__ void k_f32_to_bf16(const float* __restrict__ s, __bf16* __restrict__ d,
                              size_t n) {
  size_t i = (size_t)blockIdx.x * blockDim.x + threadIdx.x;
  if (i < n) d[i] = f2bf(s[i]);
}

// ---------------- NT GEMM: C[M,N] = A[M,K] * B[N,K]^T (+resid) ----------------
// 128x128 tile / block, BK=32, 8 waves each computing a 64x32 sub-tile (4x2 WMMAs).
// TDM path: double-buffered tensor_load_to_lds issued by wave 0, compute overlapped,
// synchronized with s_wait_tensorcnt + workgroup barrier.
__global__ __launch_bounds__(256) void k_gemm_nt(const __bf16* __restrict__ A,
                                                 const __bf16* __restrict__ Bw,
                                                 const float* __restrict__ resid,
                                                 float* __restrict__ Cf,
                                                 __bf16* __restrict__ Cb,
                                                 int M, int N, int K) {
#if HAVE_TDM
  __shared__ __bf16 sA[2][128][40];
  __shared__ __bf16 sB[2][128][40];
#else
  __shared__ __bf16 sA[1][128][40];
  __shared__ __bf16 sB[1][128][40];
#endif
  const int tid = threadIdx.x, wave = tid >> 5, lane = tid & 31;
  const int wm = wave & 1, wn = wave >> 1;            // 2x4 wave grid
  const int m0 = blockIdx.y * 128, n0 = blockIdx.x * 128;

  v8f acc[4][2] = {};

#if HAVE_TDM
  const __bf16* gA = A + (size_t)m0 * K;
  const __bf16* gB = Bw + (size_t)n0 * K;
  const unsigned ldsA[2] = {(unsigned)(size_t)&sA[0][0][0], (unsigned)(size_t)&sA[1][0][0]};
  const unsigned ldsB[2] = {(unsigned)(size_t)&sB[0][0][0], (unsigned)(size_t)&sB[1][0][0]};
  if (wave == 0) {
    tdm_load_tile(gA, ldsA[0], K);
    tdm_load_tile(gB, ldsB[0], K);
    __builtin_amdgcn_s_wait_tensorcnt(0);
  }
  __syncthreads();
  int cur = 0;
  for (int k0 = 0; k0 < K; k0 += 32) {
    if (wave == 0 && k0 + 32 < K) {       // kick off next tile into the other buffer
      tdm_load_tile(gA + k0 + 32, ldsA[cur ^ 1], K);
      tdm_load_tile(gB + k0 + 32, ldsB[cur ^ 1], K);
      __builtin_prefetch(gA + k0 + 512, 0, 1);
    }
    const __bf16* a0 = &sA[cur][0][0];
    const __bf16* b0 = &sB[cur][0][0];
    v16bf af[4], bfr[2];
#pragma unroll
    for (int i = 0; i < 4; ++i) af[i] = frag_a(a0, 40, wm * 64 + i * 16, 0, lane);
#pragma unroll
    for (int j = 0; j < 2; ++j) bfr[j] = frag_b(b0, 40, wn * 32 + j * 16, 0, lane);
#pragma unroll
    for (int i = 0; i < 4; ++i)
#pragma unroll
      for (int j = 0; j < 2; ++j) acc[i][j] = wmma_bf16(af[i], bfr[j], acc[i][j]);
    if (wave == 0) __builtin_amdgcn_s_wait_tensorcnt(0);
    __syncthreads();
    cur ^= 1;
  }
#else  // manual cooperative staging fallback
  const int lr = tid >> 1, lc = (tid & 1) * 16;
  const __bf16* ga = A + (size_t)(m0 + lr) * K + lc;
  const __bf16* gb = Bw + (size_t)(n0 + lr) * K + lc;
  for (int k0 = 0; k0 < K; k0 += 32) {
    *(v4u*)&sA[0][lr][lc]     = *(const v4u*)(ga + k0);
    *(v4u*)&sA[0][lr][lc + 8] = *(const v4u*)(ga + k0 + 8);
    *(v4u*)&sB[0][lr][lc]     = *(const v4u*)(gb + k0);
    *(v4u*)&sB[0][lr][lc + 8] = *(const v4u*)(gb + k0 + 8);
    __builtin_prefetch(ga + k0 + 256, 0, 1);
    __builtin_prefetch(gb + k0 + 256, 0, 1);
    __syncthreads();
    v16bf af[4], bfr[2];
#pragma unroll
    for (int i = 0; i < 4; ++i) af[i] = frag_a(&sA[0][0][0], 40, wm * 64 + i * 16, 0, lane);
#pragma unroll
    for (int j = 0; j < 2; ++j) bfr[j] = frag_b(&sB[0][0][0], 40, wn * 32 + j * 16, 0, lane);
#pragma unroll
    for (int i = 0; i < 4; ++i)
#pragma unroll
      for (int j = 0; j < 2; ++j) acc[i][j] = wmma_bf16(af[i], bfr[j], acc[i][j]);
    __syncthreads();
  }
#endif

  // epilogue: C element (r, lane): m = r + 8*(lane>=16), n = lane&15
  const int hi = (lane >> 4) << 3, nn = lane & 15;
  const int mbase = m0 + wm * 64 + hi, nbase = n0 + wn * 32 + nn;
#pragma unroll
  for (int i = 0; i < 4; ++i)
#pragma unroll
    for (int j = 0; j < 2; ++j) {
      const size_t base = (size_t)(mbase + i * 16) * N + (nbase + j * 16);
      if (Cf) {
        float* cp = Cf + base;
        if (resid) {
          const float* rp = resid + base;
#pragma unroll
          for (int r = 0; r < 8; ++r) cp[(size_t)r * N] = acc[i][j][r] + rp[(size_t)r * N];
        } else {
#pragma unroll
          for (int r = 0; r < 8; ++r) cp[(size_t)r * N] = acc[i][j][r];
        }
      } else {
        __bf16* cp = Cb + base;
#pragma unroll
        for (int r = 0; r < 8; ++r) cp[(size_t)r * N] = f2bf(acc[i][j][r]);
      }
    }
}

// ---------------- RoPE + (B,S,H,HD) -> (B,H,S,HD) bf16 transpose ----------------
__global__ void k_rope(const __bf16* __restrict__ qt, const __bf16* __restrict__ kt,
                       const __bf16* __restrict__ vt, const float* __restrict__ cs,
                       const float* __restrict__ sn, __bf16* __restrict__ qr,
                       __bf16* __restrict__ kr, __bf16* __restrict__ vr) {
  size_t idx = (size_t)blockIdx.x * blockDim.x + threadIdx.x;
  const size_t total = (size_t)B_ * S_ * H_ * (HD_ / 2);
  if (idx >= total) return;
  const int d = idx & 63;  size_t t = idx >> 6;     // HD/2 = 64
  const int h = t & (H_ - 1); t >>= 4;
  const int s = t & (S_ - 1); const int b = (int)(t >> 11);
  const size_t src = ((size_t)b * S_ + s) * D_ + (size_t)h * HD_ + 2 * d;
  const size_t dst = (((size_t)b * H_ + h) * S_ + s) * HD_ + 2 * d;
  const float c = cs[(size_t)s * (HD_ / 2) + d];
  const float si = sn[(size_t)s * (HD_ / 2) + d];
  float xr = bf2f(qt[src]), xi = bf2f(qt[src + 1]);
  qr[dst] = f2bf(xr * c - xi * si);
  qr[dst + 1] = f2bf(xr * si + xi * c);
  xr = bf2f(kt[src]); xi = bf2f(kt[src + 1]);
  kr[dst] = f2bf(xr * c - xi * si);
  kr[dst + 1] = f2bf(xr * si + xi * c);
  vr[dst] = vt[src];
  vr[dst + 1] = vt[src + 1];
}

// ---------------- fused causal flash attention (all-WMMA) ----------------
// grid (S/128, H, B); 8 waves, each owns 16 query rows; key tiles of 32.
__global__ __launch_bounds__(256) void k_flash_attn(const __bf16* __restrict__ Q,
                                                    const __bf16* __restrict__ Kc,
                                                    const __bf16* __restrict__ Vc,
                                                    __bf16* __restrict__ O) {
  __shared__ __bf16 sK[32][136];      // [key][hd]  (B frag for Q*K^T)
  __shared__ __bf16 sVt[128][40];     // [hd][key]  (B frag for P*V)
  __shared__ __bf16 sP[8][16][40];    // per-wave P lane-layout shuffle scratch
  const int tid = threadIdx.x, wave = tid >> 5, lane = tid & 31;
  const int b = blockIdx.z, h = blockIdx.y, q0 = blockIdx.x * 128;
  const size_t bh = ((size_t)b * H_ + h) * (size_t)S_ * HD_;
  const __bf16* Qb = Q + bh; const __bf16* Kb = Kc + bh; const __bf16* Vb = Vc + bh;
  const int qrow0 = q0 + wave * 16;

  v16bf qf[4];
#pragma unroll
  for (int t = 0; t < 4; ++t) qf[t] = frag_a(Qb, HD_, qrow0, t * 32, lane);

  v8f oacc[8] = {};
  float rmax[8], rsum[8];
#pragma unroll
  for (int r = 0; r < 8; ++r) { rmax[r] = -3.0e38f; rsum[r] = 0.f; }

  const float scale = 0.08838834764831845f;  // 1/sqrt(128)
  const int nTiles = (q0 + 128) / 32;        // causal: keys up to block's last query
  __bf16* sPw = &sP[wave][0][0];

  for (int kt = 0; kt < nTiles; ++kt) {
    const int key0 = kt * 32;
    {  // cooperative load: K tile as-is, V tile transposed to [hd][key]
      const int r = tid >> 3, c = (tid & 7) * 16;
      const __bf16* gk = Kb + (size_t)(key0 + r) * HD_ + c;
      *(v4u*)&sK[r][c] = *(const v4u*)gk;
      *(v4u*)&sK[r][c + 8] = *(const v4u*)(gk + 8);
      const __bf16* gv = Vb + (size_t)(key0 + r) * HD_ + c;
      union { v4u u[2]; __bf16 e[16]; } vv;
      vv.u[0] = *(const v4u*)gv;
      vv.u[1] = *(const v4u*)(gv + 8);
#pragma unroll
      for (int j = 0; j < 16; ++j) sVt[c + j][r] = vv.e[j];
    }
    __syncthreads();

    // S = Q * K^T : two 16-key column tiles, HD reduced in 4 WMMA steps
    v8f sc[2] = {};
#pragma unroll
    for (int j = 0; j < 2; ++j)
#pragma unroll
      for (int t = 0; t < 4; ++t) {
        v16bf kb = frag_b(&sK[0][0], 136, j * 16, t * 32, lane);
        sc[j] = wmma_bf16(qf[t], kb, sc[j]);
      }

    // online softmax per row; C-layout: row m = r + 8*(lane>=16), col n = lane&15
#pragma unroll
    for (int r = 0; r < 8; ++r) {
      const int qm = qrow0 + r + ((lane >> 4) << 3);
      const int kc = key0 + (lane & 15);
      float s0 = sc[0][r] * scale;
      float s1 = sc[1][r] * scale;
      if (kc > qm) s0 = -3.0e38f;
      if (kc + 16 > qm) s1 = -3.0e38f;
      float mij = fmaxf(s0, s1);
#pragma unroll
      for (int dx = 8; dx >= 1; dx >>= 1) mij = fmaxf(mij, __shfl_xor(mij, dx, 32));
      const float mnew = fmaxf(rmax[r], mij);
      const float corr = __expf(rmax[r] - mnew);
      rmax[r] = mnew;
      const float p0 = __expf(s0 - mnew), p1 = __expf(s1 - mnew);
      float ps = p0 + p1;
#pragma unroll
      for (int dx = 8; dx >= 1; dx >>= 1) ps += __shfl_xor(ps, dx, 32);
      rsum[r] = rsum[r] * corr + ps;
#pragma unroll
      for (int nt = 0; nt < 8; ++nt) oacc[nt][r] = oacc[nt][r] * corr;
      const int m = r + ((lane >> 4) << 3);
      sPw[m * 40 + (lane & 15)] = f2bf(p0);          // re-layout P via LDS
      sPw[m * 40 + 16 + (lane & 15)] = f2bf(p1);
    }

    // O += P(16x32) * V(32x128)
    const v16bf pa = frag_a(sPw, 40, 0, 0, lane);
#pragma unroll
    for (int nt = 0; nt < 8; ++nt) {
      v16bf vb = frag_b(&sVt[0][0], 40, nt * 16, 0, lane);
      oacc[nt] = wmma_bf16(pa, vb, oacc[nt]);
    }
    __syncthreads();
  }

  // normalize + store bf16 to (B,S,D) for the O-projection GEMM
#pragma unroll
  for (int nt = 0; nt < 8; ++nt)
#pragma unroll
    for (int r = 0; r < 8; ++r) {
      const int qm = qrow0 + r + ((lane >> 4) << 3);
      const int col = h * HD_ + nt * 16 + (lane & 15);
      O[((size_t)b * S_ + qm) * D_ + col] = f2bf(oacc[nt][r] / rsum[r]);
    }
}

// ---------------- SwiGLU: t = silu(g) * u ----------------
__global__ void k_silu_mul(const __bf16* __restrict__ g, const __bf16* __restrict__ u,
                           __bf16* __restrict__ t, size_t n) {
  size_t i = (size_t)blockIdx.x * blockDim.x + threadIdx.x;
  if (i < n) {
    const float x = bf2f(g[i]);
    t[i] = f2bf((x / (1.0f + __expf(-x))) * bf2f(u[i]));
  }
}

// ---------------- orchestration ----------------
extern "C" void kernel_launch(void* const* d_in, const int* in_sizes, int n_in,
                              void* d_out, int out_size, void* d_ws, size_t ws_size,
                              hipStream_t stream) {
  (void)in_sizes; (void)n_in; (void)out_size; (void)ws_size;
  const float* x   = (const float*)d_in[0];
  const float* fc  = (const float*)d_in[1];
  const float* fs  = (const float*)d_in[2];
  // d_in[3] = mask (causality applied analytically), d_in[13] = start_pos (==0)
  const float* wq  = (const float*)d_in[4];
  const float* wk  = (const float*)d_in[5];
  const float* wv  = (const float*)d_in[6];
  const float* wo  = (const float*)d_in[7];
  const float* w1  = (const float*)d_in[8];
  const float* w2  = (const float*)d_in[9];
  const float* w3  = (const float*)d_in[10];
  const float* anw = (const float*)d_in[11];
  const float* fnw = (const float*)d_in[12];
  float* out = (float*)d_out;

  char* ws = (char*)d_ws;
  size_t off = 0;
  auto carve = [&](size_t bytes) -> char* {
    char* p = ws + off;
    off += (bytes + 255) & ~(size_t)255;
    return p;
  };
  const size_t DD = (size_t)D_ * D_, HDd = (size_t)HID_ * D_;
  __bf16* wq_b = (__bf16*)carve(DD * 2);
  __bf16* wk_b = (__bf16*)carve(DD * 2);
  __bf16* wv_b = (__bf16*)carve(DD * 2);
  __bf16* wo_b = (__bf16*)carve(DD * 2);
  __bf16* w1_b = (__bf16*)carve(HDd * 2);
  __bf16* w3_b = (__bf16*)carve(HDd * 2);
  __bf16* w2_b = (__bf16*)carve(HDd * 2);
  const size_t MD = (size_t)M_ * D_, MH = (size_t)M_ * HID_;
  __bf16* xa_b = (__bf16*)carve(MD * 2);
  __bf16* qt = (__bf16*)carve(MD * 2);
  __bf16* kt = (__bf16*)carve(MD * 2);
  __bf16* vt = (__bf16*)carve(MD * 2);
  __bf16* qr = (__bf16*)carve(MD * 2);
  __bf16* kr = (__bf16*)carve(MD * 2);
  __bf16* vr = (__bf16*)carve(MD * 2);
  __bf16* ob = (__bf16*)carve(MD * 2);
  float*  hb = (float*)carve(MD * 4);
  __bf16* hf_b = (__bf16*)carve(MD * 2);
  __bf16* g_b = (__bf16*)carve(MH * 2);
  __bf16* u_b = (__bf16*)carve(MH * 2);
  __bf16* t_b = (__bf16*)carve(MH * 2);

  auto cvt = [&](const float* s, __bf16* d, size_t n) {
    k_f32_to_bf16<<<(unsigned)((n + 255) / 256), 256, 0, stream>>>(s, d, n);
  };
  cvt(wq, wq_b, DD); cvt(wk, wk_b, DD); cvt(wv, wv_b, DD); cvt(wo, wo_b, DD);
  cvt(w1, w1_b, HDd); cvt(w3, w3_b, HDd); cvt(w2, w2_b, HDd);

  // 1) attention RMSNorm
  k_rmsnorm_bf16<<<M_, 256, 0, stream>>>(x, anw, xa_b);

  // 2) QKV projections (bf16 out)
  dim3 gD(D_ / 128, M_ / 128);
  k_gemm_nt<<<gD, 256, 0, stream>>>(xa_b, wq_b, nullptr, nullptr, qt, M_, D_, D_);
  k_gemm_nt<<<gD, 256, 0, stream>>>(xa_b, wk_b, nullptr, nullptr, kt, M_, D_, D_);
  k_gemm_nt<<<gD, 256, 0, stream>>>(xa_b, wv_b, nullptr, nullptr, vt, M_, D_, D_);

  // 3) RoPE + head-major transpose
  const size_t nrope = (size_t)B_ * S_ * H_ * (HD_ / 2);
  k_rope<<<(unsigned)((nrope + 255) / 256), 256, 0, stream>>>(qt, kt, vt, fc, fs, qr, kr, vr);

  // 4) fused causal flash attention
  dim3 gFA(S_ / 128, H_, B_);
  k_flash_attn<<<gFA, 256, 0, stream>>>(qr, kr, vr, ob);

  // 5) O projection + residual -> h (fp32)
  k_gemm_nt<<<gD, 256, 0, stream>>>(ob, wo_b, x, hb, nullptr, M_, D_, D_);

  // 6) FFN RMSNorm
  k_rmsnorm_bf16<<<M_, 256, 0, stream>>>(hb, fnw, hf_b);

  // 7) FFN: g = hf@w1^T, u = hf@w3^T, t = silu(g)*u, out = h + t@w2^T
  dim3 gH(HID_ / 128, M_ / 128);
  k_gemm_nt<<<gH, 256, 0, stream>>>(hf_b, w1_b, nullptr, nullptr, g_b, M_, HID_, D_);
  k_gemm_nt<<<gH, 256, 0, stream>>>(hf_b, w3_b, nullptr, nullptr, u_b, M_, HID_, D_);
  k_silu_mul<<<(unsigned)((MH + 255) / 256), 256, 0, stream>>>(g_b, u_b, t_b, MH);
  k_gemm_nt<<<gD, 256, 0, stream>>>(t_b, w2_b, hb, out, nullptr, M_, D_, HID_);
}
